// TransformerBlock_47382079210028
// MI455X (gfx1250) — matrix operand order repaired and checked by reading, not compiled, not run on hardware
//
#include <hip/hip_runtime.h>

// ---------------------------------------------------------------------------
// Transformer block (B=8, S=1024, E=1024, H=16, DK=64) for gfx1250 (MI455X).
// f16 WMMA (v_wmma_f32_16x16x32_f16) for all GEMMs, fp32 accumulate.
// Software-pipelined, double-buffered LDS staging (global prefetch into
// registers overlaps WMMA compute; ds_store into the ping-pong buffer).
// ---------------------------------------------------------------------------

#define DI __device__ __forceinline__

typedef __attribute__((ext_vector_type(16))) _Float16 v16h;
typedef __attribute__((ext_vector_type(4)))  _Float16 v4h;
typedef __attribute__((ext_vector_type(8)))  float    v8f;
typedef __attribute__((ext_vector_type(4)))  unsigned int v4u;

union Frag16 { v4u u[2]; v16h h; };

constexpr int kB = 8, kS = 1024, kE = 1024, kH = 16, kDK = 64;
constexpr int kM = kB * kS;                 // 8192 rows

constexpr int EPI_F32 = 0, EPI_RELU_F16 = 1, EPI_QKV = 2, EPI_SCORES = 3, EPI_AOUT = 4;

struct GemmP {
  const void*      A;    long lda;  size_t bsA;   // bsA in BYTES (per-z batch stride)
  const _Float16*  B;    long ldb;  size_t bsB;   // bsB in ELEMENTS
  const float*     bias;
  void*            out;
  _Float16 *q, *k, *v;                            // EPI_QKV targets
  const unsigned char* pad;                       // EPI_SCORES padding mask [B,S]
  int M, N, K;
};

// Generic WMMA GEMM:  D[m,n] = sum_k A[m,k] * B[n,k]  (B stored [N,K] row-major)
template<int BM, int BN, int BK, int WM, int WN, int EPI, bool AF32>
__launch_bounds__(256)
__global__ void gemm_wmma(GemmP p) {
  constexpr int NWM = BM / WM, NWN = BN / WN;
  constexpr int NTH = NWM * NWN * 32;
  constexpr int LD  = BK + 8;                     // LDS row stride in halves
  constexpr int MT  = WM / 16, NTI = WN / 16;
  constexpr int SEG = BK / 8;                     // 8-half segments per row
  constexpr int APT = BM * SEG / NTH;             // A segments per thread
  constexpr int BPT = BN * SEG / NTH;             // B segments per thread

  __shared__ __align__(16) _Float16 sA[2 * BM * LD];
  __shared__ __align__(16) _Float16 sB[2 * BN * LD];

  const int tid  = threadIdx.x;
  const int wv   = tid >> 5, lane = tid & 31;
  const int wm   = wv % NWM, wn = wv / NWM;
  const int hl   = lane >> 4, mr = lane & 15;
  const int z    = blockIdx.z;
  const long m0  = (long)blockIdx.y * BM;
  const long n0  = (long)blockIdx.x * BN;

  const char*      Ab = (const char*)p.A + (size_t)z * p.bsA;
  const _Float16*  Bp = p.B + (size_t)z * p.bsB;

  v8f acc[MT][NTI] = {};

  // Per-thread global prefetch registers.
  v4u    ra[APT];
  float4 raf[2 * APT];
  v4u    rb[BPT];

  auto fetch = [&](int kb) {
#pragma unroll
    for (int j = 0; j < APT; ++j) {
      const int i = tid + j * NTH;
      const int row = i / SEG, sg = i % SEG;
      if (AF32) {
        const float* s = (const float*)Ab + (size_t)(m0 + row) * p.lda + kb + sg * 8;
        raf[2 * j]     = ((const float4*)s)[0];
        raf[2 * j + 1] = ((const float4*)s)[1];
      } else {
        ra[j] = *(const v4u*)((const _Float16*)Ab + (size_t)(m0 + row) * p.lda + kb + sg * 8);
      }
    }
#pragma unroll
    for (int j = 0; j < BPT; ++j) {
      const int i = tid + j * NTH;
      const int row = i / SEG, sg = i % SEG;
      rb[j] = *(const v4u*)&Bp[(size_t)(n0 + row) * p.ldb + kb + sg * 8];
    }
  };

  auto stage = [&](int buf) {
    _Float16* dA = &sA[buf * BM * LD];
    _Float16* dB = &sB[buf * BN * LD];
#pragma unroll
    for (int j = 0; j < APT; ++j) {
      const int i = tid + j * NTH;
      const int row = i / SEG, sg = i % SEG;
      _Float16* dst = &dA[row * LD + sg * 8];
      if (AF32) {
        float4 f0 = raf[2 * j], f1 = raf[2 * j + 1];
        v4h h0 = { (_Float16)f0.x, (_Float16)f0.y, (_Float16)f0.z, (_Float16)f0.w };
        v4h h1 = { (_Float16)f1.x, (_Float16)f1.y, (_Float16)f1.z, (_Float16)f1.w };
        *(v4h*)dst       = h0;
        *(v4h*)(dst + 4) = h1;
      } else {
        *(v4u*)dst = ra[j];
      }
    }
#pragma unroll
    for (int j = 0; j < BPT; ++j) {
      const int i = tid + j * NTH;
      const int row = i / SEG, sg = i % SEG;
      *(v4u*)&dB[row * LD + sg * 8] = rb[j];
    }
  };

  auto compute = [&](int buf) {
    const _Float16* cA = &sA[buf * BM * LD];
    const _Float16* cB = &sB[buf * BN * LD];
#pragma unroll
    for (int kk = 0; kk < BK; kk += 32) {
      Frag16 af[MT], bf[NTI];
#pragma unroll
      for (int tm = 0; tm < MT; ++tm) {
        // A layout (16-bit 16x32): lane m=mr; K(e) = kk + e + (e&8) + 8*hl
        const _Float16* b_ = &cA[(wm * WM + tm * 16 + mr) * LD + kk + 8 * hl];
        af[tm].u[0] = *(const v4u*)(b_);        // K = kk + 8*hl + 0..7
        af[tm].u[1] = *(const v4u*)(b_ + 16);   // K = kk + 8*hl + 16..23
      }
#pragma unroll
      for (int tn = 0; tn < NTI; ++tn) {
        // B layout (16-bit 32x16): lane n=mr; K(e) = kk + e + (e&8) + 8*hl
        const _Float16* b_ = &cB[(wn * WN + tn * 16 + mr) * LD + kk + 8 * hl];
        bf[tn].u[0] = *(const v4u*)(b_);
        bf[tn].u[1] = *(const v4u*)(b_ + 16);
      }
#pragma unroll
      for (int tm = 0; tm < MT; ++tm)
#pragma unroll
        for (int tn = 0; tn < NTI; ++tn)
          acc[tm][tn] = __builtin_amdgcn_wmma_f32_16x16x32_f16(
              false, af[tm].h, false, bf[tn].h, (short)0, acc[tm][tn], false, false);
    }
  };

  // ---- software pipeline: fetch(t+1) | compute(t) | stage(t+1) | barrier ----
  fetch(0);
  stage(0);
  __syncthreads();
  for (int step = 0;; ++step) {
    const bool more = (step + 1) * BK < p.K;
    if (more) fetch((step + 1) * BK);
    compute(step & 1);
    if (!more) break;
    stage((step + 1) & 1);
    __syncthreads();
  }

  // ---- epilogue: D layout — m = r + 8*hl, n = mr (per 16x16 tile) ----
#pragma unroll
  for (int tm = 0; tm < MT; ++tm) {
#pragma unroll
    for (int tn = 0; tn < NTI; ++tn) {
      const long n = n0 + wn * WN + tn * 16 + mr;
      float bv = p.bias ? p.bias[n] : 0.f;
#pragma unroll
      for (int r = 0; r < 8; ++r) {
        const long m = m0 + wm * WM + tm * 16 + r + 8 * hl;
        float val = acc[tm][tn][r] + bv;
        if (EPI == EPI_F32) {
          ((float*)p.out)[(size_t)m * p.N + n] = val;
        } else if (EPI == EPI_RELU_F16) {
          ((_Float16*)p.out)[(size_t)m * p.N + n] = (_Float16)fmaxf(val, 0.f);
        } else if (EPI == EPI_QKV) {
          // qkv row m = b*S+s, col n in [0,3E): part|head|dk
          int part = (int)(n >> 10), w = (int)(n & 1023);
          int hh = w >> 6, dk = w & 63;
          int b = (int)(m >> 10), s = (int)(m & 1023);
          _Float16 hv = (_Float16)val;
          if (part == 0)      p.q[(((size_t)b * kH + hh) * kS + s) * kDK + dk] = hv;
          else if (part == 1) p.k[(((size_t)b * kH + hh) * kS + s) * kDK + dk] = hv;
          else                p.v[(((size_t)b * kH + hh) * kDK + dk) * kS + s] = hv; // V^T
        } else if (EPI == EPI_SCORES) {
          // clip(s/8) -> causal/pad -10000 -> clip  ==  masked entries are exactly -10
          float s = fminf(fmaxf(val * 0.125f, -10.f), 10.f);
          int b = z >> 4;                                   // z = b*H + h
          if (n > m || p.pad[(size_t)b * kS + n]) s = -10.f;
          ((float*)p.out)[(size_t)z * kS * kS + (size_t)m * kS + n] = s;
        } else if (EPI == EPI_AOUT) {
          int b = z >> 4, hh = z & 15;
          ((_Float16*)p.out)[((size_t)b * kS + m) * kE + hh * kDK + n] = (_Float16)val;
        }
      }
    }
  }
}

// ---------------------------------------------------------------------------
// Block reductions (256 threads = 8 waves, wave32)
// ---------------------------------------------------------------------------
DI float wave_sum(float v) {
#pragma unroll
  for (int o = 16; o > 0; o >>= 1) v += __shfl_xor(v, o);
  return v;
}
DI float wave_max(float v) {
#pragma unroll
  for (int o = 16; o > 0; o >>= 1) v = fmaxf(v, __shfl_xor(v, o));
  return v;
}
DI float blk_sum(float v, float* sh) {
  v = wave_sum(v);
  if ((threadIdx.x & 31) == 0) sh[threadIdx.x >> 5] = v;
  __syncthreads();
  if (threadIdx.x == 0) {
    float s = 0.f;
    for (int i = 0; i < 8; ++i) s += sh[i];
    sh[8] = s;
  }
  __syncthreads();
  float r = sh[8];
  __syncthreads();
  return r;
}
DI float blk_max(float v, float* sh) {
  v = wave_max(v);
  if ((threadIdx.x & 31) == 0) sh[threadIdx.x >> 5] = v;
  __syncthreads();
  if (threadIdx.x == 0) {
    float s = sh[0];
    for (int i = 1; i < 8; ++i) s = fmaxf(s, sh[i]);
    sh[8] = s;
  }
  __syncthreads();
  float r = sh[8];
  __syncthreads();
  return r;
}

// Softmax over one 1024-wide score row, in place, with pad-zero + renormalize.
__launch_bounds__(256)
__global__ void softmax_rows(float* attn, const unsigned char* __restrict__ pad) {
  __shared__ float sh[9];
  const size_t row = blockIdx.x;              // row = (b*H + h)*S + q
  const int b = (int)(row >> 14);             // / (H*S)
  float* pr = attn + row * (size_t)kS;
  const unsigned char* pd = pad + (size_t)b * kS;
  const int t = threadIdx.x;

  float4 v = ((const float4*)pr)[t];
  float mx = blk_max(fmaxf(fmaxf(v.x, v.y), fmaxf(v.z, v.w)), sh);
  float e0 = __expf(v.x - mx), e1 = __expf(v.y - mx);
  float e2 = __expf(v.z - mx), e3 = __expf(v.w - mx);
  uchar4 pm = ((const uchar4*)pd)[t];
  float sa = blk_sum(e0 + e1 + e2 + e3, sh);
  float sn = blk_sum((pm.x ? 0.f : e0) + (pm.y ? 0.f : e1) +
                     (pm.z ? 0.f : e2) + (pm.w ? 0.f : e3), sh);
  float wsum = fmaxf(sn / sa, 1e-8f);
  float coef = 1.f / (sa * wsum);
  float4 o;
  o.x = pm.x ? 0.f : e0 * coef;
  o.y = pm.y ? 0.f : e1 * coef;
  o.z = pm.z ? 0.f : e2 * coef;
  o.w = pm.w ? 0.f : e3 * coef;
  ((float4*)pr)[t] = o;
}

// out = LayerNorm(x + nan_to_num(y)) * g + b ; optional f16 mirror for next GEMM.
__launch_bounds__(256)
__global__ void add_ln(const float* __restrict__ x, const float* __restrict__ y,
                       const float* __restrict__ g, const float* __restrict__ bt,
                       float* __restrict__ out, _Float16* __restrict__ outh) {
  __shared__ float sh[9];
  const size_t row = blockIdx.x;
  const int t = threadIdx.x;
  float4 xv = ((const float4*)(x + row * (size_t)kE))[t];
  float4 yv = ((const float4*)(y + row * (size_t)kE))[t];
  float tv[4] = { xv.x, xv.y, xv.z, xv.w };
  float yy[4] = { yv.x, yv.y, yv.z, yv.w };
#pragma unroll
  for (int i = 0; i < 4; ++i) tv[i] += __builtin_isfinite(yy[i]) ? yy[i] : 0.f;
  float mu = blk_sum(tv[0] + tv[1] + tv[2] + tv[3], sh) * (1.f / kE);
  float d0 = tv[0] - mu, d1 = tv[1] - mu, d2 = tv[2] - mu, d3 = tv[3] - mu;
  float var = blk_sum(d0 * d0 + d1 * d1 + d2 * d2 + d3 * d3, sh) * (1.f / kE);
  float rs = rsqrtf(var + 1e-5f);
  float4 gv = ((const float4*)g)[t];
  float4 bv = ((const float4*)bt)[t];
  float4 o;
  o.x = d0 * rs * gv.x + bv.x;
  o.y = d1 * rs * gv.y + bv.y;
  o.z = d2 * rs * gv.z + bv.z;
  o.w = d3 * rs * gv.w + bv.w;
  ((float4*)(out + row * (size_t)kE))[t] = o;
  if (outh) {
    v4h hv = { (_Float16)o.x, (_Float16)o.y, (_Float16)o.z, (_Float16)o.w };
    ((v4h*)(outh + row * (size_t)kE))[t] = hv;
  }
}

__launch_bounds__(256)
__global__ void cvt_f32_to_f16(const float* __restrict__ s, _Float16* __restrict__ d, long n4) {
  long i = blockIdx.x * (long)blockDim.x + threadIdx.x;
  long stride = (long)gridDim.x * blockDim.x;
  for (; i < n4; i += stride) {
    float4 f = ((const float4*)s)[i];
    v4h h = { (_Float16)f.x, (_Float16)f.y, (_Float16)f.z, (_Float16)f.w };
    ((v4h*)d)[i] = h;
  }
}

// ---------------------------------------------------------------------------
extern "C" void kernel_launch(void* const* d_in, const int* in_sizes, int n_in,
                              void* d_out, int out_size, void* d_ws, size_t ws_size,
                              hipStream_t stream) {
  (void)in_sizes; (void)n_in; (void)out_size; (void)ws_size;

  const float*         x_in   = (const float*)d_in[0];
  const unsigned char* padm   = (const unsigned char*)d_in[1];   // bool [B,S]
  /* d_in[2] causal_mask: analytically triu(k=1) — not read */
  const float* qkv_w = (const float*)d_in[3];
  const float* qkv_b = (const float*)d_in[4];
  const float* out_w = (const float*)d_in[5];
  const float* out_b = (const float*)d_in[6];
  const float* ff1_w = (const float*)d_in[7];
  const float* ff1_b = (const float*)d_in[8];
  const float* ff2_w = (const float*)d_in[9];
  const float* ff2_b = (const float*)d_in[10];
  const float* ln1_g = (const float*)d_in[11];
  const float* ln1_b = (const float*)d_in[12];
  const float* ln2_g = (const float*)d_in[13];
  const float* ln2_b = (const float*)d_in[14];

  float* out_x = (float*)d_out;                                  // [B,S,E]
  float* attn  = (float*)d_out + (size_t)kM * kE;                // [B,H,S,S]

  // workspace layout
  char* wp = (char*)d_ws;
  auto alloc = [&](size_t bytes) -> char* {
    char* r = wp; wp += (bytes + 255) & ~(size_t)255; return r;
  };
  _Float16* xh   = (_Float16*)alloc((size_t)kM * kE * 2);        // f16 x
  _Float16* qh   = (_Float16*)alloc((size_t)kB * kH * kS * kDK * 2);
  _Float16* kh   = (_Float16*)alloc((size_t)kB * kH * kS * kDK * 2);
  _Float16* vT   = (_Float16*)alloc((size_t)kB * kH * kDK * kS * 2); // V^T [B,H,DK,S]
  float*    raw  = (float*)   alloc((size_t)kM * kE * 4);        // GEMM fp32 out (reused)
  float*    x1   = (float*)   alloc((size_t)kM * kE * 4);        // LN1 result
  _Float16* hbuf = (_Float16*)alloc((size_t)kM * 4 * kE * 2);    // relu(ff1)
  _Float16* wqkv = (_Float16*)alloc((size_t)3 * kE * kE * 2);
  _Float16* wout = (_Float16*)alloc((size_t)kE * kE * 2);
  _Float16* wff1 = (_Float16*)alloc((size_t)4 * kE * kE * 2);
  _Float16* wff2 = (_Float16*)alloc((size_t)4 * kE * kE * 2);
  _Float16* acat = xh;                                           // reuse (xh dead post-QKV)
  _Float16* x1h  = qh;                                           // reuse (qh dead post-scores)

  // 0) fp32 -> f16 conversions
  cvt_f32_to_f16<<<2048, 256, 0, stream>>>(x_in,  xh,   (long)kM * kE / 4);
  cvt_f32_to_f16<<<1024, 256, 0, stream>>>(qkv_w, wqkv, (long)3 * kE * kE / 4);
  cvt_f32_to_f16<<<512,  256, 0, stream>>>(out_w, wout, (long)kE * kE / 4);
  cvt_f32_to_f16<<<1024, 256, 0, stream>>>(ff1_w, wff1, (long)4 * kE * kE / 4);
  cvt_f32_to_f16<<<1024, 256, 0, stream>>>(ff2_w, wff2, (long)4 * kE * kE / 4);

  GemmP p{};

  // 1) QKV projection: [8192,1024] x [3072,1024]^T -> scatter q/k/v^T (f16)
  p.A = xh; p.lda = kE; p.bsA = 0;
  p.B = wqkv; p.ldb = kE; p.bsB = 0;
  p.bias = qkv_b; p.q = qh; p.k = kh; p.v = vT;
  p.M = kM; p.N = 3 * kE; p.K = kE;
  gemm_wmma<128,128,64,64,32,EPI_QKV,false>
      <<<dim3(3 * kE / 128, kM / 128, 1), 256, 0, stream>>>(p);

  // 2) Scores: per (b,h): Q[1024,64] x K[1024,64]^T -> masked/clipped fp32 into d_out
  p = GemmP{};
  p.A = qh; p.lda = kDK; p.bsA = (size_t)kS * kDK * 2;
  p.B = kh; p.ldb = kDK; p.bsB = (size_t)kS * kDK;
  p.out = attn; p.pad = padm;
  p.M = kS; p.N = kS; p.K = kDK;
  gemm_wmma<128,128,64,64,32,EPI_SCORES,false>
      <<<dim3(kS / 128, kS / 128, kB * kH), 256, 0, stream>>>(p);

  // 3) Softmax rows (in place in d_out), pad-zero + renormalize
  softmax_rows<<<dim3((unsigned)((size_t)kB * kH * kS)), 256, 0, stream>>>(attn, padm);

  // 4) attn_out = P @ V : per (b,h): [1024,1024](fp32) x V^T[64,1024] -> acat f16 [B,S,E]
  p = GemmP{};
  p.A = attn; p.lda = kS; p.bsA = (size_t)kS * kS * 4;
  p.B = vT; p.ldb = kS; p.bsB = (size_t)kDK * kS;
  p.out = acat;
  p.M = kS; p.N = kDK; p.K = kS;
  gemm_wmma<128,64,64,32,32,EPI_AOUT,true>
      <<<dim3(1, kS / 128, kB * kH), 256, 0, stream>>>(p);

  // 5) Output projection -> raw fp32
  p = GemmP{};
  p.A = acat; p.lda = kE; p.bsA = 0;
  p.B = wout; p.ldb = kE; p.bsB = 0;
  p.bias = out_b; p.out = raw;
  p.M = kM; p.N = kE; p.K = kE;
  gemm_wmma<128,128,64,64,32,EPI_F32,false>
      <<<dim3(kE / 128, kM / 128, 1), 256, 0, stream>>>(p);

  // 6) x1 = LN(x + nan_to_num(attn_out))
  add_ln<<<kM, 256, 0, stream>>>(x_in, raw, ln1_g, ln1_b, x1, x1h);

  // 7) h = relu(x1 @ ff1_w^T + b) -> f16
  p = GemmP{};
  p.A = x1h; p.lda = kE; p.bsA = 0;
  p.B = wff1; p.ldb = kE; p.bsB = 0;
  p.bias = ff1_b; p.out = hbuf;
  p.M = kM; p.N = 4 * kE; p.K = kE;
  gemm_wmma<128,128,64,64,32,EPI_RELU_F16,false>
      <<<dim3(4 * kE / 128, kM / 128, 1), 256, 0, stream>>>(p);

  // 8) ff = h @ ff2_w^T + b -> raw fp32
  p = GemmP{};
  p.A = hbuf; p.lda = 4 * kE; p.bsA = 0;
  p.B = wff2; p.ldb = 4 * kE; p.bsB = 0;
  p.bias = ff2_b; p.out = raw;
  p.M = kM; p.N = kE; p.K = 4 * kE;
  gemm_wmma<128,128,64,64,32,EPI_F32,false>
      <<<dim3(kE / 128, kM / 128, 1), 256, 0, stream>>>(p);

  // 9) out = LN(x1 + nan_to_num(ff)) -> d_out x-region
  add_ln<<<kM, 256, 0, stream>>>(x1, raw, ln2_g, ln2_b, out_x, nullptr);
}
